// DCMN_71408126263909
// MI455X (gfx1250) — matrix operand order repaired and checked by reading, not compiled, not run on hardware
//
#include <hip/hip_runtime.h>

typedef __bf16 bf16;
typedef __attribute__((ext_vector_type(16))) __bf16 v16bf;
typedef __attribute__((ext_vector_type(8)))  __bf16 v8bf;
typedef __attribute__((ext_vector_type(8)))  float  v8f;

static constexpr int cB = 32;   // batch
static constexpr int cN = 32;   // passages
static constexpr int cP = 64;   // tokens per passage
static constexpr int cQ = 32;   // query tokens
static constexpr int cA = 24;   // tokens per answer
static constexpr int cM = 4;    // answers
static constexpr int cL = 768;  // hidden
static constexpr int cK = 2;    // top-k passages
static constexpr int cMA = cM * cA;  // 96
static constexpr int cXP = cK * cP;  // 128

#define FINF __builtin_inff()

__device__ __forceinline__ bf16 f2bf(float x) {
  union { float f; unsigned u; } v; v.f = x;
  unsigned r = v.u + 0x7fffu + ((v.u >> 16) & 1u);
  union { unsigned short s; bf16 b; } o;
  o.s = (unsigned short)(r >> 16);
  return o.b;
}

// ---------------------------------------------------------------------------
// Generic batched WMMA GEMM: C[b] = act(A[b] (Mr x Kr, row-major bf16)
//                                     x  B[b]^T with B given as Nr rows x Kr)
// Kr must be a multiple of 32. Writes f32 (Cf) and/or bf16 (Cb) outputs.
// One 16x32 tile per wave (2 N-tiles share the A fragment -> 2 WMMAs/k-step);
// out-of-range rows read a zero row so the k-loop has no divergence and EXEC
// stays all-ones. 8 waves per block (wave32).
// ---------------------------------------------------------------------------
__global__ __launch_bounds__(256) void wmma_gemm(
    const bf16* __restrict__ A, long sA, int lda,
    const bf16* __restrict__ Bt, long sB, int ldb,
    const bf16* __restrict__ zrow,
    float* __restrict__ Cf, bf16* __restrict__ Cb, long sC, int ldc,
    int Mr, int Nr, int Kr, int relu)
{
  const int batch = blockIdx.z;
  const bf16* Ab = A + (size_t)batch * sA;
  const bf16* Bb = Bt + (size_t)batch * sB;
  const int Mt  = (Mr + 15) >> 4;
  const int Nt  = (Nr + 15) >> 4;
  const int Nt2 = (Nt + 1) >> 1;
  const int wave = blockIdx.x * (blockDim.x >> 5) + (threadIdx.x >> 5);
  if (wave >= Mt * Nt2) return;
  const int mt = wave / Nt2, np = wave % Nt2;
  const int lane = threadIdx.x & 31;
  const int sel  = lane >> 4;
  const int m  = mt * 16 + (lane & 15);
  const int n0 = np * 32 + (lane & 15);
  const int n1 = n0 + 16;
  const bf16* arow  = (m  < Mr) ? Ab + (size_t)m  * lda : zrow;
  const bf16* brow0 = (n0 < Nr) ? Bb + (size_t)n0 * ldb : zrow;
  const bf16* brow1 = (n1 < Nr) ? Bb + (size_t)n1 * ldb : zrow;

  v8f acc0 = {0.f, 0.f, 0.f, 0.f, 0.f, 0.f, 0.f, 0.f};
  v8f acc1 = {0.f, 0.f, 0.f, 0.f, 0.f, 0.f, 0.f, 0.f};
  for (int k0 = 0; k0 < Kr; k0 += 32) {
    v16bf af, b0, b1;
    {
      // A 16x32 bf16 layout: lanes 0-15 K[0..15]; lanes 16-31 K[16..31]
      v8bf lo = *(const v8bf*)(arow + k0 + sel * 16);
      v8bf hi = *(const v8bf*)(arow + k0 + sel * 16 + 8);
#pragma unroll
      for (int i = 0; i < 8; ++i) { af[i] = lo[i]; af[i + 8] = hi[i]; }
    }
    {
      // B 32x16 bf16 layout: lanes 0-15 K[0..15]; lanes 16-31 K[16..31]
      v8bf lo = *(const v8bf*)(brow0 + k0 + sel * 16);
      v8bf hi = *(const v8bf*)(brow0 + k0 + sel * 16 + 8);
#pragma unroll
      for (int i = 0; i < 8; ++i) { b0[i] = lo[i]; b0[i + 8] = hi[i]; }
    }
    {
      v8bf lo = *(const v8bf*)(brow1 + k0 + sel * 16);
      v8bf hi = *(const v8bf*)(brow1 + k0 + sel * 16 + 8);
#pragma unroll
      for (int i = 0; i < 8; ++i) { b1[i] = lo[i]; b1[i + 8] = hi[i]; }
    }
    // stream ahead of the WMMA pipeline (global_prefetch_b8)
    __builtin_prefetch(arow + k0 + 128, 0, 1);
    __builtin_prefetch(brow0 + k0 + 128, 0, 1);
    __builtin_prefetch(brow1 + k0 + 128, 0, 1);
    acc0 = __builtin_amdgcn_wmma_f32_16x16x32_bf16(false, af, false, b0,
                                                   (short)0, acc0, false, false);
    acc1 = __builtin_amdgcn_wmma_f32_16x16x32_bf16(false, af, false, b1,
                                                   (short)0, acc1, false, false);
  }
  {
    const int mbase = mt * 16 + sel * 8;
    float* cf = Cf ? Cf + (size_t)batch * sC : nullptr;
    bf16*  cb = Cb ? Cb + (size_t)batch * sC : nullptr;
#pragma unroll
    for (int r = 0; r < 8; ++r) {
      int mm = mbase + r;
      if (mm < Mr) {
        float v0 = acc0[r], v1 = acc1[r];
        if (relu) { v0 = fmaxf(v0, 0.f); v1 = fmaxf(v1, 0.f); }
        if (n0 < Nr) {
          if (cf) cf[(size_t)mm * ldc + n0] = v0;
          if (cb) cb[(size_t)mm * ldc + n0] = f2bf(v0);
        }
        if (n1 < Nr) {
          if (cf) cf[(size_t)mm * ldc + n1] = v1;
          if (cb) cb[(size_t)mm * ldc + n1] = f2bf(v1);
        }
      }
    }
  }
}

// ---------------------------------------------------------------------------
// Fused score kernel: per (b,n) compute col-max of
//   Pn(64x768) x [An | Qn]^T (768x128), normalization fused via 1/||row||.
// 8 waves = 8 column tiles; 4 row tiles accumulated into running col-max.
// ---------------------------------------------------------------------------
__global__ __launch_bounds__(256) void score_k(
    const float* __restrict__ Hp, const float* __restrict__ Ha,
    const float* __restrict__ Hq, const float* __restrict__ rnP,
    const float* __restrict__ rnA, const float* __restrict__ rnQ,
    float* __restrict__ score)
{
  const int bn = blockIdx.x;
  const int b  = bn >> 5;
  const int wv   = threadIdx.x >> 5;
  const int lane = threadIdx.x & 31;
  const int sel  = lane >> 4;
  const int col  = wv * 16 + (lane & 15);

  const float* Ab  = Hp  + (size_t)bn * cP * cL;
  const float* rnp = rnP + (size_t)bn * cP;

  const float* brow; float bscale;
  if (col < cMA) {
    brow = Ha + ((size_t)b * cMA + col) * cL;
    bscale = rnA[b * cMA + col];
  } else {
    brow = Hq + ((size_t)b * cQ + (col - cMA)) * cL;
    bscale = rnQ[b * cQ + (col - cMA)];
  }

  float cmax = -FINF;
  for (int mt = 0; mt < 4; ++mt) {
    const int m = mt * 16 + (lane & 15);
    const float* arow = Ab + (size_t)m * cL;
    const float ascale = rnp[m];
    v8f acc = {0.f, 0.f, 0.f, 0.f, 0.f, 0.f, 0.f, 0.f};
    for (int k0 = 0; k0 < cL; k0 += 32) {
      v16bf af, bfr;
      {
        const float* p = arow + k0 + sel * 16;
#pragma unroll
        for (int i = 0; i < 8; ++i) af[i] = f2bf(p[i] * ascale);
        p += 8;
#pragma unroll
        for (int i = 0; i < 8; ++i) af[i + 8] = f2bf(p[i] * ascale);
      }
      {
        const float* p = brow + k0 + sel * 16;
#pragma unroll
        for (int i = 0; i < 16; ++i) bfr[i] = f2bf(p[i] * bscale);
      }
      __builtin_prefetch(arow + k0 + 64, 0, 1);
      __builtin_prefetch(brow + k0 + 64, 0, 1);
      acc = __builtin_amdgcn_wmma_f32_16x16x32_bf16(false, af, false, bfr,
                                                    (short)0, acc, false, false);
    }
#pragma unroll
    for (int i = 0; i < 8; ++i) cmax = fmaxf(cmax, acc[i]);
  }
  // lane l and l^16 hold the same column -> combine for full 16-row max
  cmax = fmaxf(cmax, __shfl_xor(cmax, 16, 32));
  __shared__ float sc[128];
  if (lane < 16) sc[wv * 16 + lane] = cmax;
  __syncthreads();
  if (threadIdx.x == 0) {
    float sa = 0.f, sq = 0.f;
    for (int c = 0; c < cMA; ++c) sa += sc[c];
    for (int c = cMA; c < 128; ++c) sq += sc[c];
    score[bn] = sa * (1.f / (float)cA) + sq * (1.f / (float)cQ);
  }
}

// ---------------------------------------------------------------------------
// Small helpers
// ---------------------------------------------------------------------------
__global__ void fill0_k(bf16* __restrict__ d, int n)
{
  int i = blockIdx.x * blockDim.x + threadIdx.x;
  if (i < n) d[i] = f2bf(0.f);
}

__global__ __launch_bounds__(256) void rownorm_k(const float* __restrict__ X,
                                                 float* __restrict__ rn)
{
  const size_t row = blockIdx.x;
  const float* p = X + row * cL;
  float s = 0.f;
  for (int i = threadIdx.x; i < cL; i += 256) { float v = p[i]; s = fmaf(v, v, s); }
  __shared__ float sm[256];
  sm[threadIdx.x] = s; __syncthreads();
  for (int o = 128; o > 0; o >>= 1) {
    if (threadIdx.x < o) sm[threadIdx.x] += sm[threadIdx.x + o];
    __syncthreads();
  }
  if (threadIdx.x == 0) rn[row] = 1.f / fmaxf(sqrtf(sm[0]), 1e-8f);
}

__global__ void cast_k(const float* __restrict__ s, bf16* __restrict__ d, long n)
{
  long i = (long)blockIdx.x * blockDim.x + threadIdx.x;
  if (i < n) d[i] = f2bf(s[i]);
}

// dst[c*R + r] = src[r*C + c]  (bf16 transpose-cast of weights)
__global__ void tcast_k(const float* __restrict__ src, bf16* __restrict__ dst,
                        int R, int C)
{
  long i = (long)blockIdx.x * blockDim.x + threadIdx.x;
  if (i < (long)R * C) {
    int r = (int)(i / C), c = (int)(i % C);
    dst[(size_t)c * R + r] = f2bf(src[i]);
  }
}

__global__ __launch_bounds__(32) void topk_k(const float* __restrict__ score,
                                             int* __restrict__ idx)
{
  const int b = blockIdx.x;
  __shared__ float s[cN];
  s[threadIdx.x] = score[b * cN + threadIdx.x];
  __syncthreads();
  if (threadIdx.x == 0) {
    int i0 = 0;
    for (int c = 1; c < cN; ++c) if (s[c] > s[i0]) i0 = c;
    s[i0] = -FINF;
    int i1 = 0;
    for (int c = 1; c < cN; ++c) if (s[c] > s[i1]) i1 = c;
    idx[b * cK + 0] = i0;
    idx[b * cK + 1] = i1;
  }
}

__global__ __launch_bounds__(256) void gather_k(const float* __restrict__ Hp,
                                                const int* __restrict__ idx,
                                                float* __restrict__ Hps,
                                                bf16* __restrict__ HpsB)
{
  const int row = blockIdx.x;                 // b * 128 + k * 64 + p
  const int b = row / cXP;
  const int r = row % cXP;
  const int k = r / cP;
  const int p = r % cP;
  const int n = idx[b * cK + k];
  const float* src = Hp + (((size_t)b * cN + n) * cP + p) * cL;
  float* df = Hps + (size_t)row * cL;
  bf16*  db = HpsB + (size_t)row * cL;
  for (int c = threadIdx.x; c < cL; c += 256) {
    float v = src[c];
    df[c] = v; db[c] = f2bf(v);
  }
}

// softmax over rows of length len (<=128), in place
__global__ __launch_bounds__(128) void softmax_k(float* __restrict__ D, int ld, int len)
{
  float* p = D + (size_t)blockIdx.x * ld;
  const int t = threadIdx.x;
  __shared__ float sm[128];
  float v = (t < len) ? p[t] : -FINF;
  sm[t] = v; __syncthreads();
  for (int o = 64; o > 0; o >>= 1) { if (t < o) sm[t] = fmaxf(sm[t], sm[t + o]); __syncthreads(); }
  float mx = sm[0]; __syncthreads();
  float e = (t < len) ? __expf(v - mx) : 0.f;
  sm[t] = e; __syncthreads();
  for (int o = 64; o > 0; o >>= 1) { if (t < o) sm[t] += sm[t + o]; __syncthreads(); }
  if (t < len) p[t] = e / sm[0];
}

// C[b,m,:768] = act( sum_{k<Kr} G[b,m,k] * H[b,k,:768] ), bf16 output
__global__ __launch_bounds__(256) void small_gemm(
    const float* __restrict__ G, long sG, int ldg,
    const float* __restrict__ H, long sH, int ldh,
    bf16* __restrict__ Cb, long sC, int ldc, int Kr, int relu)
{
  const int b = blockIdx.z;
  const int m = blockIdx.x;
  __shared__ float gs[128];
  const float* gr = G + (size_t)b * sG + (size_t)m * ldg;
  if (threadIdx.x < Kr) gs[threadIdx.x] = gr[threadIdx.x];
  __syncthreads();
  const float* Hb = H + (size_t)b * sH;
  bf16* cr = Cb + (size_t)b * sC + (size_t)m * ldc;
  for (int n = threadIdx.x; n < cL; n += 256) {
    float s = 0.f;
    for (int k = 0; k < Kr; ++k) s = fmaf(gs[k], Hb[(size_t)k * ldh + n], s);
    if (relu) s = fmaxf(s, 0.f);
    cr[n] = f2bf(s);
  }
}

__global__ __launch_bounds__(256) void colmax_k(const float* __restrict__ S, long sS,
                                                int rows, float* __restrict__ Df,
                                                bf16* __restrict__ Db)
{
  const int b = blockIdx.z;
  const int n = blockIdx.x * blockDim.x + threadIdx.x;
  if (n >= cL) return;
  const float* p = S + (size_t)b * sS + n;
  float mx = -FINF;
  for (int r = 0; r < rows; ++r) mx = fmaxf(mx, p[(size_t)r * cL]);
  Df[(size_t)b * cL + n] = mx;
  Db[(size_t)b * cL + n] = f2bf(mx);
}

__global__ void gate_ho_k(const float* __restrict__ t7, const float* __restrict__ t8,
                          const float* __restrict__ b8, const float* __restrict__ Hi,
                          const float* __restrict__ bar, float* __restrict__ Ho,
                          bf16* __restrict__ HoB, long n)
{
  long i = (long)blockIdx.x * blockDim.x + threadIdx.x;
  if (i >= n) return;
  int l = (int)(i % cL);
  float g = 1.f / (1.f + __expf(-(t7[i] + t8[i] + b8[l])));
  float v = g * Hi[i] + (1.f - g) * bar[i];
  Ho[i] = v; HoB[i] = f2bf(v);
}

__global__ void gate_match_k(const float* __restrict__ gx, const float* __restrict__ gy,
                             const float* __restrict__ b14, const float* __restrict__ Sxy,
                             const float* __restrict__ Syx, float* __restrict__ out, long n)
{
  long i = (long)blockIdx.x * blockDim.x + threadIdx.x;
  if (i >= n) return;
  int l = (int)(i % cL);
  float g = 1.f / (1.f + __expf(-(gx[i] + gy[i] + b14[l])));
  out[i] = g * Sxy[i] + (1.f - g) * Syx[i];
}

__global__ __launch_bounds__(128) void final_k(const float* __restrict__ Mpq,
                                               const float* __restrict__ Mpo,
                                               const float* __restrict__ Mqo,
                                               const float* __restrict__ V,
                                               float* __restrict__ out)
{
  const int b = blockIdx.x;
  const int wv = threadIdx.x >> 5, lane = threadIdx.x & 31;
  const float* a0 = Mpq + (size_t)b * cL;
  const float* a1 = Mpo + ((size_t)wv * cB + b) * cL;
  const float* a2 = Mqo + ((size_t)wv * cB + b) * cL;
  float s = 0.f;
  for (int l = lane; l < cL; l += 32)
    s += a0[l] * V[l] + a1[l] * V[cL + l] + a2[l] * V[2 * cL + l];
  for (int o = 16; o > 0; o >>= 1) s += __shfl_xor(s, o, 32);
  __shared__ float lg[cM];
  if (lane == 0) lg[wv] = s;
  __syncthreads();
  if (threadIdx.x == 0) {
    float mx = lg[0];
    for (int i = 1; i < cM; ++i) mx = fmaxf(mx, lg[i]);
    float se = 0.f;
    for (int i = 0; i < cM; ++i) se += __expf(lg[i] - mx);
    float lse = mx + __logf(se);
    for (int i = 0; i < cM; ++i) out[b * cM + i] = lg[i] - lse;
  }
}

// ---------------------------------------------------------------------------
// Host orchestration
// ---------------------------------------------------------------------------
extern "C" void kernel_launch(void* const* d_in, const int* in_sizes, int n_in,
                              void* d_out, int out_size, void* d_ws, size_t ws_size,
                              hipStream_t stream)
{
  (void)in_sizes; (void)n_in; (void)out_size; (void)ws_size;
  const float* Hp  = (const float*)d_in[0];
  const float* Hq  = (const float*)d_in[1];
  const float* Ha  = (const float*)d_in[2];
  const float* W5  = (const float*)d_in[3];
  const float* W6  = (const float*)d_in[4];
  const float* W7  = (const float*)d_in[5];
  const float* W8  = (const float*)d_in[6];
  const float* b8  = (const float*)d_in[7];
  const float* W9  = (const float*)d_in[8];
  const float* W10 = (const float*)d_in[9];
  const float* W11 = (const float*)d_in[10];
  const float* W12 = (const float*)d_in[11];
  const float* W13 = (const float*)d_in[12];
  const float* W14 = (const float*)d_in[13];
  const float* b14 = (const float*)d_in[14];
  const float* Vp  = (const float*)d_in[15];

  char* base = (char*)d_ws;
  size_t off = 0;
  auto alloc = [&](size_t bytes) -> void* {
    void* p = base + off;
    off += (bytes + 255) & ~(size_t)255;
    return p;
  };

  const size_t LL = (size_t)cL * cL;
  bf16* zbuf = (bf16*)alloc((size_t)4096 * 2);   // zero row for GEMM edge tiles
  bf16* W5t  = (bf16*)alloc(LL * 2);
  bf16* W7t  = (bf16*)alloc(LL * 2);
  bf16* W8t  = (bf16*)alloc(LL * 2);
  bf16* W9t  = (bf16*)alloc(LL * 2);
  bf16* W10t = (bf16*)alloc(LL * 2);
  bf16* W11t = (bf16*)alloc(LL * 2);
  bf16* W12t = (bf16*)alloc(LL * 2);
  bf16* W13t = (bf16*)alloc(LL * 2);
  bf16* W14t = (bf16*)alloc(LL * 2);
  bf16* W6t  = (bf16*)alloc((size_t)3 * LL * 2);

  float* rnP = (float*)alloc((size_t)cB * cN * cP * 4);
  float* rnA = (float*)alloc((size_t)cB * cMA * 4);
  float* rnQ = (float*)alloc((size_t)cB * cQ * 4);
  bf16*  HaB = (bf16*)alloc((size_t)cB * cMA * cL * 2);
  bf16*  HqB = (bf16*)alloc((size_t)cB * cQ * cL * 2);
  float* score = (float*)alloc((size_t)cB * cN * 4);
  int*   idx   = (int*)alloc((size_t)cB * cK * 4);
  float* Hps  = (float*)alloc((size_t)cB * cXP * cL * 4);
  bf16*  HpsB = (bf16*)alloc((size_t)cB * cXP * cL * 2);

  bf16*  c1oB  = (bf16*)alloc((size_t)cB * cM * cA * cL * 2);
  float* Gho   = (float*)alloc((size_t)cB * cA * cA * 4);
  bf16*  concB = (bf16*)alloc((size_t)cB * cA * 3 * cL * 2);
  float* barF  = (float*)alloc((size_t)cB * cM * cA * cL * 4);
  bf16*  barB  = (bf16*)alloc((size_t)cB * cM * cA * cL * 2);
  float* t7    = (float*)alloc((size_t)cB * cM * cA * cL * 4);
  float* t8    = (float*)alloc((size_t)cB * cM * cA * cL * 4);
  float* HoF   = (float*)alloc((size_t)cB * cM * cA * cL * 4);
  bf16*  HoB   = (bf16*)alloc((size_t)cB * cM * cA * cL * 2);

  bf16* mc1B = (bf16*)alloc((size_t)cB * cXP * cL * 2);
  bf16* mc2B = (bf16*)alloc((size_t)cB * cQ * cL * 2);
  bf16* mc3B = (bf16*)alloc((size_t)cB * cQ * cL * 2);
  bf16* mc4B = (bf16*)alloc((size_t)cB * cM * cA * cL * 2);

  float* logXY = (float*)alloc((size_t)cB * cXP * cQ * 4);
  float* logYX = (float*)alloc((size_t)cB * cQ * cXP * 4);
  bf16*  ExB   = (bf16*)alloc((size_t)cB * cXP * cL * 2);
  bf16*  EyB   = (bf16*)alloc((size_t)cB * cQ * cL * 2);
  float* SxyF  = (float*)alloc((size_t)cB * cXP * cL * 4);
  float* SyxF  = (float*)alloc((size_t)cB * cQ * cL * 4);
  float* Sxy   = (float*)alloc((size_t)cB * cL * 4);
  bf16*  SxyB  = (bf16*)alloc((size_t)cB * cL * 2);
  float* Syx   = (float*)alloc((size_t)cB * cL * 4);
  bf16*  SyxB  = (bf16*)alloc((size_t)cB * cL * 2);
  float* gx    = (float*)alloc((size_t)cB * cL * 4);
  float* gy    = (float*)alloc((size_t)cB * cL * 4);
  float* Mpq   = (float*)alloc((size_t)cB * cL * 4);
  float* Mpo   = (float*)alloc((size_t)cM * cB * cL * 4);
  float* Mqo   = (float*)alloc((size_t)cM * cB * cL * 4);

  auto launch_gemm = [&](const bf16* Ap, long sA, int lda, const bf16* Bp, long sB,
                         int ldb, float* Cf, bf16* Cb, long sC, int ldc,
                         int Mr, int Nr, int Kr, int relu, int batch) {
    int Mt = (Mr + 15) / 16, Nt2 = ((Nr + 15) / 16 + 1) / 2;
    int blocks = (Mt * Nt2 + 7) / 8;
    wmma_gemm<<<dim3(blocks, 1, batch), dim3(256), 0, stream>>>(
        Ap, sA, lda, Bp, sB, ldb, zbuf, Cf, Cb, sC, ldc, Mr, Nr, Kr, relu);
  };
  auto launch_small = [&](const float* G, long sG, int ldg, const float* H, long sH,
                          int ldh, bf16* Cb, long sC, int ldc, int Mr, int Kr,
                          int relu, int batch) {
    small_gemm<<<dim3(Mr, 1, batch), dim3(256), 0, stream>>>(
        G, sG, ldg, H, sH, ldh, Cb, sC, ldc, Kr, relu);
  };
  auto launch_softmax = [&](float* D, int rows, int len) {
    softmax_k<<<dim3(rows), dim3(128), 0, stream>>>(D, len, len);
  };
  auto launch_colmax = [&](const float* S, long sS, int rows, float* Df, bf16* Db) {
    colmax_k<<<dim3(3, 1, cB), dim3(256), 0, stream>>>(S, sS, rows, Df, Db);
  };
  auto launch_cast = [&](const float* s, bf16* d, long n) {
    cast_k<<<dim3((unsigned)((n + 255) / 256)), dim3(256), 0, stream>>>(s, d, n);
  };
  auto launch_tcast = [&](const float* s, bf16* d, int R, int C) {
    long n = (long)R * C;
    tcast_k<<<dim3((unsigned)((n + 255) / 256)), dim3(256), 0, stream>>>(s, d, R, C);
  };

  // ---- zero row (must precede all WMMA GEMMs) ----
  fill0_k<<<dim3(16), dim3(256), 0, stream>>>(zbuf, 4096);

  // ---- weight transposition / casting ----
  launch_tcast(W5, W5t, cL, cL);
  launch_tcast(W7, W7t, cL, cL);
  launch_tcast(W8, W8t, cL, cL);
  launch_tcast(W9, W9t, cL, cL);
  launch_tcast(W10, W10t, cL, cL);
  launch_tcast(W11, W11t, cL, cL);
  launch_tcast(W12, W12t, cL, cL);
  launch_tcast(W13, W13t, cL, cL);
  launch_tcast(W14, W14t, cL, cL);
  launch_tcast(W6, W6t, 3 * cL, cL);

  // ---- norms + raw bf16 casts ----
  rownorm_k<<<dim3(cB * cN * cP), dim3(256), 0, stream>>>(Hp, rnP);
  rownorm_k<<<dim3(cB * cMA), dim3(256), 0, stream>>>(Ha, rnA);
  rownorm_k<<<dim3(cB * cQ), dim3(256), 0, stream>>>(Hq, rnQ);
  launch_cast(Ha, HaB, (long)cB * cMA * cL);
  launch_cast(Hq, HqB, (long)cB * cQ * cL);

  // ---- passage scoring (fused normalize + WMMA + col-max) + top-k gather ----
  score_k<<<dim3(cB * cN), dim3(256), 0, stream>>>(Hp, Ha, Hq, rnP, rnA, rnQ, score);
  topk_k<<<dim3(cB), dim3(32), 0, stream>>>(score, idx);
  gather_k<<<dim3(cB * cXP), dim3(256), 0, stream>>>(Hp, idx, Hps, HpsB);

  // ---- H_o stage ----
  launch_gemm(HaB, (long)cA * cL, cL, W5t, 0, cL, nullptr, c1oB,
              (long)cA * cL, cL, cA, cL, cL, 0, cB * cM);
  for (int i = 0; i < cM; ++i) {
    int slot = 0;
    for (int j = 0; j < cM; ++j) {
      if (j == i) continue;
      launch_gemm(c1oB + (size_t)i * cA * cL, (long)cM * cA * cL, cL,
                  HaB + (size_t)j * cA * cL, (long)cM * cA * cL, cL,
                  Gho, nullptr, (long)cA * cA, cA, cA, cA, cL, 0, cB);
      launch_softmax(Gho, cB * cA, cA);
      launch_small(Gho, (long)cA * cA, cA, Ha + (size_t)j * cA * cL,
                   (long)cM * cA * cL, cL, concB + (size_t)slot * cL,
                   (long)cA * 3 * cL, 3 * cL, cA, cA, 1, cB);
      ++slot;
    }
    launch_gemm(concB, (long)cA * 3 * cL, 3 * cL, W6t, 0, 3 * cL,
                barF + (size_t)i * cA * cL, barB + (size_t)i * cA * cL,
                (long)cM * cA * cL, cL, cA, cL, 3 * cL, 0, cB);
  }
  launch_gemm(barB, (long)cA * cL, cL, W7t, 0, cL, t7, nullptr,
              (long)cA * cL, cL, cA, cL, cL, 0, cB * cM);
  launch_gemm(barB, (long)cA * cL, cL, W8t, 0, cL, t8, nullptr,
              (long)cA * cL, cL, cA, cL, cL, 0, cB * cM);
  {
    long n = (long)cB * cM * cA * cL;
    gate_ho_k<<<dim3((unsigned)((n + 255) / 256)), dim3(256), 0, stream>>>(
        t7, t8, b8, Ha, barF, HoF, HoB, n);
  }

  // ---- match caches ----
  launch_gemm(HpsB, (long)cXP * cL, cL, W9t, 0, cL, nullptr, mc1B,
              (long)cXP * cL, cL, cXP, cL, cL, 0, cB);
  launch_gemm(HqB, (long)cQ * cL, cL, W9t, 0, cL, nullptr, mc2B,
              (long)cQ * cL, cL, cQ, cL, cL, 0, cB);
  launch_gemm(HqB, (long)cQ * cL, cL, W10t, 0, cL, nullptr, mc3B,
              (long)cQ * cL, cL, cQ, cL, cL, 0, cB);
  launch_gemm(HoB, (long)cA * cL, cL, W10t, 0, cL, nullptr, mc4B,
              (long)cA * cL, cL, cA, cL, cL, 0, cB * cM);

  auto run_match = [&](int X, int Y,
                       const bf16* GLx, long sGLx,
                       const bf16* HyBp, long sHy, const float* HyFp, long sHyF,
                       const bf16* GLy, long sGLy,
                       const bf16* HxBp, long sHx, const float* HxFp, long sHxF,
                       float* Mout) {
    launch_gemm(GLx, sGLx, cL, HyBp, sHy, cL, logXY, nullptr, (long)X * Y, Y,
                X, Y, cL, 0, cB);
    launch_softmax(logXY, cB * X, Y);
    launch_gemm(GLy, sGLy, cL, HxBp, sHx, cL, logYX, nullptr, (long)Y * X, X,
                Y, X, cL, 0, cB);
    launch_softmax(logYX, cB * Y, X);
    launch_small(logXY, (long)X * Y, Y, HyFp, sHyF, cL, ExB, (long)X * cL, cL,
                 X, Y, 0, cB);
    launch_small(logYX, (long)Y * X, X, HxFp, sHxF, cL, EyB, (long)Y * cL, cL,
                 Y, X, 0, cB);
    launch_gemm(ExB, (long)X * cL, cL, W11t, 0, cL, SxyF, nullptr,
                (long)X * cL, cL, X, cL, cL, 1, cB);
    launch_gemm(EyB, (long)Y * cL, cL, W12t, 0, cL, SyxF, nullptr,
                (long)Y * cL, cL, Y, cL, cL, 1, cB);
    launch_colmax(SxyF, (long)X * cL, X, Sxy, SxyB);
    launch_colmax(SyxF, (long)Y * cL, Y, Syx, SyxB);
    // stacked (B x L) gemv as single M=32 WMMA GEMM
    launch_gemm(SxyB, 0, cL, W13t, 0, cL, gx, nullptr, 0, cL, cB, cL, cL, 0, 1);
    launch_gemm(SyxB, 0, cL, W14t, 0, cL, gy, nullptr, 0, cL, cB, cL, cL, 0, 1);
    long n = (long)cB * cL;
    gate_match_k<<<dim3((unsigned)((n + 255) / 256)), dim3(256), 0, stream>>>(
        gx, gy, b14, Sxy, Syx, Mout, n);
  };

  run_match(cXP, cQ,
            mc1B, (long)cXP * cL,
            HqB, (long)cQ * cL, Hq, (long)cQ * cL,
            mc3B, (long)cQ * cL,
            HpsB, (long)cXP * cL, Hps, (long)cXP * cL,
            Mpq);
  for (int i = 0; i < cM; ++i)
    run_match(cXP, cA,
              mc1B, (long)cXP * cL,
              HoB + (size_t)i * cA * cL, (long)cM * cA * cL,
              HoF + (size_t)i * cA * cL, (long)cM * cA * cL,
              mc4B + (size_t)i * cA * cL, (long)cM * cA * cL,
              HpsB, (long)cXP * cL, Hps, (long)cXP * cL,
              Mpo + (size_t)i * cB * cL);
  for (int i = 0; i < cM; ++i)
    run_match(cQ, cA,
              mc2B, (long)cQ * cL,
              HoB + (size_t)i * cA * cL, (long)cM * cA * cL,
              HoF + (size_t)i * cA * cL, (long)cM * cA * cL,
              mc4B + (size_t)i * cA * cL, (long)cM * cA * cL,
              HqB, (long)cQ * cL, Hq, (long)cQ * cL,
              Mqo + (size_t)i * cB * cL);

  final_k<<<dim3(cB), dim3(128), 0, stream>>>(Mpq, Mpo, Mqo, Vp, (float*)d_out);
}
